// BigBirdBlock_69020124446907
// MI455X (gfx1250) — matrix operand order repaired and checked by reading, not compiled, not run on hardware
//
#include <hip/hip_runtime.h>
#include <hip/hip_bf16.h>

// ---------------------------------------------------------------------------
// Problem constants (match reference)
// ---------------------------------------------------------------------------
constexpr int kB  = 4;
constexpr int kS  = 4096;
constexpr int kD  = 512;
constexpr int kH  = 8;
constexpr int kDH = 64;
constexpr int kM  = 2048;
constexpr int kNB = 64;          // sequence blocks (S / 64)
constexpr int kRows = kB * kS;   // 16384

typedef __bf16 bf16;
typedef __attribute__((ext_vector_type(16))) __bf16 v16bf;
typedef __attribute__((ext_vector_type(8)))  __bf16 v8bf;
typedef __attribute__((ext_vector_type(8)))  float  v8f;
typedef __attribute__((ext_vector_type(4)))  int    v4i;

// Address-space-qualified pointer types for the CDNA5 async/TDM builtins.
typedef __attribute__((address_space(1))) v4i  gv4i;   // global (HIP __device__)
typedef __attribute__((address_space(3))) v4i  lv4i;   // LDS
typedef __attribute__((address_space(3))) bf16 l_bf16;

// ---------------------------------------------------------------------------
// CDNA5 async global->LDS staging (ASYNCcnt) with portable fallback.
// Signature (from hipcc diagnostic): (v4i as1* src, v4i as3* dst, Ii, Ii).
// ---------------------------------------------------------------------------
#if __has_builtin(__builtin_amdgcn_global_load_async_to_lds_b128)
#define HAVE_ASYNC_LDS 1
#else
#define HAVE_ASYNC_LDS 0
#endif

static __device__ __forceinline__ void cp16(bf16* dst_lds, const bf16* src_glb) {
#if HAVE_ASYNC_LDS
    __builtin_amdgcn_global_load_async_to_lds_b128(
        (gv4i*)src_glb, (lv4i*)dst_lds, /*offset=*/0, /*cpol=*/0);
#else
    *(v8bf*)dst_lds = *(const v8bf*)src_glb;
#endif
}

static __device__ __forceinline__ void async_wait() {
#if HAVE_ASYNC_LDS
#if __has_builtin(__builtin_amdgcn_s_wait_asynccnt)
    __builtin_amdgcn_s_wait_asynccnt(0);
#else
    asm volatile("s_wait_asynccnt 0x0" ::: "memory");
#endif
#endif
}

// ---------------------------------------------------------------------------
// Tensor Data Mover: one-shot DMA of a contiguous 8KB (2048 dword) tile
// from global memory into LDS.  D# packed per CDNA5 ISA 8.3/8.4:
//   group0: count=1, lds_addr, global_addr[56:0], type=2
//   group1: data_size=4B, tensor_dim0=2048, tensor_dim1=1,
//           tile_dim0=2048, tile_dim1=1, dim0_stride=2048
// 6-arg builtin form (per hipcc diagnostic): g0, g1, g2, g3, g4, cpol.
// Issued by a single wave; completion via TENSORcnt.
// ---------------------------------------------------------------------------
#if __has_builtin(__builtin_amdgcn_tensor_load_to_lds)
#define HAVE_TDM 1
typedef unsigned int u32x4 __attribute__((ext_vector_type(4)));
typedef int i32x4 __attribute__((ext_vector_type(4)));
typedef int i32x8 __attribute__((ext_vector_type(8)));

static __device__ __forceinline__ void tdm_load_8k(bf16* lds_dst, const bf16* gsrc) {
    const unsigned lds_off = (unsigned)(size_t)(l_bf16*)lds_dst;
    const unsigned long long ga = (unsigned long long)(size_t)gsrc;
    u32x4 g0;
    g0[0] = 1u;                                            // count=1
    g0[1] = lds_off;                                       // lds_addr
    g0[2] = (unsigned)ga;                                  // global_addr[31:0]
    g0[3] = ((unsigned)(ga >> 32) & 0x01FFFFFFu) | (2u << 30);  // [56:32] | type=2
    i32x8 g1;
    g1[0] = (int)(2u << 16);      // workgroup_mask=0, data_size=2 (4B units)
    g1[1] = (int)(2048u << 16);   // tensor_dim0 = 2048 (low16 in [31:16])
    g1[2] = (int)(1u << 16);      // tensor_dim0 hi=0, tensor_dim1 = 1
    g1[3] = (int)(2048u << 16);   // tensor_dim1 hi=0, tile_dim0 = 2048
    g1[4] = 1;                    // tile_dim1 = 1
    g1[5] = 2048;                 // tensor_dim0_stride = 2048
    g1[6] = 0;
    g1[7] = 0;
    const i32x4 gz4 = (i32x4)0;
    const i32x8 gz8 = (i32x8)0;
    __builtin_amdgcn_tensor_load_to_lds(g0, g1, gz4, gz4, gz8, /*cpol=*/0);
}

static __device__ __forceinline__ void tdm_wait() {
    __builtin_amdgcn_s_wait_tensorcnt(0);
}
#else
#define HAVE_TDM 0
#endif

// ---------------------------------------------------------------------------
// bf16 helpers (bit-exact RNE convert)
// ---------------------------------------------------------------------------
static __device__ __forceinline__ bf16 f2bf(float f) {
    union { float f; unsigned u; } x; x.f = f;
    unsigned r = x.u + 0x7fffu + ((x.u >> 16) & 1u);
    union { unsigned short s; bf16 b; } y; y.s = (unsigned short)(r >> 16);
    return y.b;
}

// ---------------------------------------------------------------------------
// WMMA fragment loads from LDS, per CDNA5 ISA 7.12.2 layouts (wave32).
// ---------------------------------------------------------------------------
union FragU { v16bf v; v8bf h[2]; };

static __device__ __forceinline__ v16bf load_afrag(const bf16* base, int row0,
                                                   int k0, int ld, int lane) {
    const int r  = row0 + (lane & 15);
    const int kb = k0 + ((lane >> 4) << 3);
    FragU u;
    u.h[0] = *(const v8bf*)(base + r * ld + kb);
    u.h[1] = *(const v8bf*)(base + r * ld + kb + 16);
    return u.v;
}

static __device__ __forceinline__ v16bf load_bfrag(const bf16* base, int n0,
                                                   int k0, int ld, int lane) {
    const int nn = n0 + (lane & 15);
    const int kb = k0 + ((lane >> 4) << 3);
    FragU u;
    u.h[0] = *(const v8bf*)(base + nn * ld + kb);
    u.h[1] = *(const v8bf*)(base + nn * ld + kb + 16);
    return u.v;
}

static __device__ __forceinline__ v8f wmma_bf16(v16bf a, v16bf b, v8f c) {
    return __builtin_amdgcn_wmma_f32_16x16x32_bf16(
        false, a, false, b, (short)0, c, false, false);
}

// ---------------------------------------------------------------------------
// LayerNorm (fp32 in) -> bf16 out.  One wave per row (D=512 -> 16 elem/lane).
// ---------------------------------------------------------------------------
__global__ __launch_bounds__(256)
void ln_bf16_kernel(const float* __restrict__ x, const float* __restrict__ sc,
                    const float* __restrict__ bi, bf16* __restrict__ out) {
    const int lane = threadIdx.x & 31;
    const int row  = blockIdx.x * 8 + (threadIdx.x >> 5);
    const float* xr = x + (size_t)row * kD;
    float vbuf[16];
    float s = 0.f;
#pragma unroll
    for (int i = 0; i < 16; ++i) { vbuf[i] = xr[lane + i * 32]; s += vbuf[i]; }
#pragma unroll
    for (int o = 16; o > 0; o >>= 1) s += __shfl_xor(s, o, 32);
    const float mu = s * (1.0f / kD);
    float var = 0.f;
#pragma unroll
    for (int i = 0; i < 16; ++i) { float d = vbuf[i] - mu; var += d * d; }
#pragma unroll
    for (int o = 16; o > 0; o >>= 1) var += __shfl_xor(var, o, 32);
    const float rs = rsqrtf(var * (1.0f / kD) + 1e-6f);
    bf16* orow = out + (size_t)row * kD;
#pragma unroll
    for (int i = 0; i < 16; ++i) {
        const int c = lane + i * 32;
        orow[c] = f2bf((vbuf[i] - mu) * rs * sc[c] + bi[c]);
    }
}

// ---------------------------------------------------------------------------
// Weight pack: fp32 [K][N] row-major -> bf16 [N][K] (WMMA B operand layout).
// ---------------------------------------------------------------------------
__global__ __launch_bounds__(256)
void pack_wT_kernel(const float* __restrict__ src, bf16* __restrict__ dst,
                    int K, int N, float scale) {
    const int i  = blockIdx.x * 256 + threadIdx.x;
    const int nn = i / K;
    const int kk = i - nn * K;
    dst[i] = f2bf(src[(size_t)kk * N + nn] * scale);
}

// ---------------------------------------------------------------------------
// BigBird connectivity table (device-built, deterministic).
// ---------------------------------------------------------------------------
__global__ void init_idx_kernel(int* __restrict__ idx, int* __restrict__ cnt) {
    const int n = threadIdx.x;
    if (n >= kNB) return;
    int* row = idx + n * 64;
    if (n == 0 || n == kNB - 1) {
        for (int j = 0; j < kNB; ++j) row[j] = j;
        cnt[n] = kNB;
        return;
    }
    int list[8];
    int c = 0;
    const int cand[5] = {0, n - 1, n, n + 1, kNB - 1};
    for (int j = 0; j < 5; ++j) {
        bool dup = false;
        for (int q = 0; q < c; ++q) dup |= (list[q] == cand[j]);
        if (!dup) list[c++] = cand[j];
    }
    unsigned state = 0x9E3779B9u * (unsigned)(n + 1) + 12345u;
    int added = 0;
    while (added < 3) {
        state = state * 1664525u + 1013904223u;
        const int bk = (int)((state >> 16) & (kNB - 1));
        bool dup = false;
        for (int q = 0; q < c; ++q) dup |= (list[q] == bk);
        if (!dup) { list[c++] = bk; ++added; }
    }
    for (int j = 0; j < c; ++j) row[j] = list[j];
    cnt[n] = c;
}

// ---------------------------------------------------------------------------
// Generic bf16 WMMA GEMM: C[M x N] = A[M x K] * Bt[N x K]^T  (f32 accum)
// Block tile 128x128, 8 waves, wave tile 32x64 (2x4 WMMA accums).
// MODE 0: QKV   -> scatter bf16 into [mat][B][H][S][DH]
// MODE 1: OPROJ -> f32 out = C + add (residual)
// MODE 2: FFN1  -> bf16 out = gelu(C + bias)
// MODE 3: FFN2  -> f32 out = C + bias + add
// ---------------------------------------------------------------------------
template <int MODE>
__global__ __launch_bounds__(256)
void gemm_bf16_kernel(const bf16* __restrict__ A, const bf16* __restrict__ Bt,
                      int N, int K,
                      bf16* __restrict__ outB, float* __restrict__ outF,
                      const float* __restrict__ add,
                      const float* __restrict__ bias) {
    __shared__ __align__(16) bf16 As[128 * 32];
    __shared__ __align__(16) bf16 Bs[128 * 32];

    const int t    = threadIdx.x;
    const int w    = t >> 5;
    const int lane = t & 31;
    const int wm   = w & 3;
    const int wn   = w >> 2;
    const int bm   = blockIdx.y;
    const int bn   = blockIdx.x;

    const bf16* Ab = A  + (size_t)bm * 128 * K;
    const bf16* Bb = Bt + (size_t)bn * 128 * K;

    const int srow  = t >> 1;
    const int shalf = t & 1;

    v8f acc[2][4];
#pragma unroll
    for (int i = 0; i < 2; ++i)
#pragma unroll
        for (int j = 0; j < 4; ++j) acc[i][j] = (v8f)0.0f;

    for (int k0 = 0; k0 < K; k0 += 32) {
        // Async-stage A and Bt slabs (128 x 32 bf16 each) into LDS.
        {
            const bf16* sa = Ab + (size_t)srow * K + k0 + shalf * 16;
            bf16* da = As + srow * 32 + shalf * 16;
            cp16(da, sa);
            cp16(da + 8, sa + 8);
            const bf16* sb = Bb + (size_t)srow * K + k0 + shalf * 16;
            bf16* db = Bs + srow * 32 + shalf * 16;
            cp16(db, sb);
            cp16(db + 8, sb + 8);
            if (k0 + 32 < K) {
                __builtin_prefetch(Ab + (size_t)srow * K + k0 + 32, 0, 1);
                __builtin_prefetch(Bb + (size_t)srow * K + k0 + 32, 0, 1);
            }
        }
        async_wait();
        __syncthreads();

        v16bf aF[2], bF[4];
        aF[0] = load_afrag(As, wm * 32,      0, 32, lane);
        aF[1] = load_afrag(As, wm * 32 + 16, 0, 32, lane);
#pragma unroll
        for (int j = 0; j < 4; ++j)
            bF[j] = load_bfrag(Bs, wn * 64 + j * 16, 0, 32, lane);
#pragma unroll
        for (int i = 0; i < 2; ++i)
#pragma unroll
            for (int j = 0; j < 4; ++j)
                acc[i][j] = wmma_bf16(aF[i], bF[j], acc[i][j]);
        __syncthreads();
    }

#pragma unroll
    for (int i = 0; i < 2; ++i) {
#pragma unroll
        for (int j = 0; j < 4; ++j) {
#pragma unroll
            for (int r = 0; r < 8; ++r) {
                const int grow = bm * 128 + wm * 32 + i * 16 + r + ((lane >> 4) << 3);
                const int gcol = bn * 128 + wn * 64 + j * 16 + (lane & 15);
                const float c = acc[i][j][r];
                if constexpr (MODE == 0) {
                    const int mat = gcol >> 9;
                    const int hh  = (gcol >> 6) & (kH - 1);
                    const int dh  = gcol & (kDH - 1);
                    const int bb  = grow >> 12;
                    const int ss  = grow & (kS - 1);
                    outB[((((size_t)mat * kB + bb) * kH + hh) * kS + ss) * kDH + dh] =
                        f2bf(c);
                } else if constexpr (MODE == 1) {
                    const size_t o = (size_t)grow * N + gcol;
                    outF[o] = c + add[o];
                } else if constexpr (MODE == 2) {
                    const float xg = c + bias[gcol];
                    const float g = 0.5f * xg *
                        (1.0f + tanhf(0.7978845608f * (xg + 0.044715f * xg * xg * xg)));
                    outB[(size_t)grow * N + gcol] = f2bf(g);
                } else {
                    const size_t o = (size_t)grow * N + gcol;
                    outF[o] = c + bias[gcol] + add[o];
                }
            }
        }
    }
}

// ---------------------------------------------------------------------------
// BigBird block-sparse attention with online softmax.
// One 256-thread block per (b, h, query-block).  K tile is DMA'd by the
// Tensor Data Mover when available; V is staged transposed for the WMMA B
// operand; Q via async-LDS copies.
// ---------------------------------------------------------------------------
__global__ __launch_bounds__(256)
void bigbird_attn_kernel(const bf16* __restrict__ q, const bf16* __restrict__ k,
                         const bf16* __restrict__ v, const int* __restrict__ idx,
                         const int* __restrict__ cnt, bf16* __restrict__ O) {
    __shared__ __align__(16) bf16 Qs[64 * 64];
    __shared__ __align__(16) bf16 Ks[64 * 64];
    __shared__ __align__(16) bf16 Ps[64 * 64];
    __shared__ __align__(16) bf16 Vts[64 * 64];   // transposed: [dh][key]
    __shared__ float Ss[64 * 64];
    __shared__ float mrow[64], lrow[64], crow[64];

    const int t    = threadIdx.x;
    const int w    = t >> 5;
    const int lane = t & 31;
    const int bid  = blockIdx.x;
    const int n    = bid & (kNB - 1);
    const int h    = (bid >> 6) & (kH - 1);
    const int b    = bid >> 9;
    const int mt   = w & 3;
    const int ntg  = w >> 2;

    const bf16* qb = q + ((((size_t)b * kH + h) * kS) + (size_t)n * 64) * kDH;
    for (int i = t; i < 512; i += 256) cp16(Qs + i * 8, qb + i * 8);
    if (t < 64) { mrow[t] = -1e30f; lrow[t] = 0.0f; }

    v8f accO[2];
    accO[0] = (v8f)0.0f;
    accO[1] = (v8f)0.0f;

    const int ecount = cnt[n];
    async_wait();
    __syncthreads();

    for (int e = 0; e < ecount; ++e) {
        const int kb = idx[n * 64 + e];
        const bf16* kbp = k + ((((size_t)b * kH + h) * kS) + (size_t)kb * 64) * kDH;
        const bf16* vbp = v + ((((size_t)b * kH + h) * kS) + (size_t)kb * 64) * kDH;
#if HAVE_TDM
        if (w == 0) tdm_load_8k(Ks, kbp);        // one wave drives the TDM
#else
        for (int i = t; i < 512; i += 256) cp16(Ks + i * 8, kbp + i * 8);
#endif
        for (int i = t; i < 4096; i += 256) {
            const int key = i >> 6, dh = i & 63;
            Vts[dh * 64 + key] = vbp[i];
        }
        async_wait();
#if HAVE_TDM
        if (w == 0) tdm_wait();
#endif
        __syncthreads();

        // --- scores: each wave computes two 16x16 tiles of S = Q K^T ---
        v8f sacc[2];
        sacc[0] = (v8f)0.0f;
        sacc[1] = (v8f)0.0f;
#pragma unroll
        for (int ks = 0; ks < 2; ++ks) {
            const v16bf aF = load_afrag(Qs, mt * 16, ks * 32, 64, lane);
            const v16bf b0 = load_bfrag(Ks, (ntg * 2 + 0) * 16, ks * 32, 64, lane);
            const v16bf b1 = load_bfrag(Ks, (ntg * 2 + 1) * 16, ks * 32, 64, lane);
            sacc[0] = wmma_bf16(aF, b0, sacc[0]);
            sacc[1] = wmma_bf16(aF, b1, sacc[1]);
        }
#pragma unroll
        for (int r = 0; r < 8; ++r) {
            const int row = mt * 16 + r + ((lane >> 4) << 3);
            Ss[row * 64 + (ntg * 2 + 0) * 16 + (lane & 15)] = sacc[0][r];
            Ss[row * 64 + (ntg * 2 + 1) * 16 + (lane & 15)] = sacc[1][r];
        }
        __syncthreads();

        // --- online softmax: thread t<64 owns query row t ---
        if (t < 64) {
            const float* srow = &Ss[t * 64];
            float mx = -1e30f;
            for (int j2 = 0; j2 < 64; ++j2) mx = fmaxf(mx, srow[j2]);
            const float mo = mrow[t];
            const float mn = fmaxf(mo, mx);
            const float cc = __expf(mo - mn);
            float sum = 0.f;
            for (int j2 = 0; j2 < 64; ++j2) {
                const float p = __expf(srow[j2] - mn);
                sum += p;
                Ps[t * 64 + j2] = f2bf(p);
            }
            mrow[t] = mn;
            lrow[t] = lrow[t] * cc + sum;
            crow[t] = cc;
        }
        __syncthreads();

        // --- rescale accumulators, then O += P * V ---
#pragma unroll
        for (int r = 0; r < 8; ++r) {
            const int row = mt * 16 + r + ((lane >> 4) << 3);
            const float f = crow[row];
            accO[0][r] *= f;
            accO[1][r] *= f;
        }
#pragma unroll
        for (int ks = 0; ks < 2; ++ks) {
            const v16bf aF = load_afrag(Ps, mt * 16, ks * 32, 64, lane);
            const v16bf b0 = load_bfrag(Vts, (ntg * 2 + 0) * 16, ks * 32, 64, lane);
            const v16bf b1 = load_bfrag(Vts, (ntg * 2 + 1) * 16, ks * 32, 64, lane);
            accO[0] = wmma_bf16(aF, b0, accO[0]);
            accO[1] = wmma_bf16(aF, b1, accO[1]);
        }
        __syncthreads();
    }

    // --- normalize by softmax denominator, write O[b][s][h*64+dh] (bf16) ---
#pragma unroll
    for (int j = 0; j < 2; ++j) {
#pragma unroll
        for (int r = 0; r < 8; ++r) {
            const int row = mt * 16 + r + ((lane >> 4) << 3);
            const int col = (ntg * 2 + j) * 16 + (lane & 15);
            const float val = accO[j][r] / lrow[row];
            const size_t s = (size_t)n * 64 + row;
            O[((size_t)b * kS + s) * kD + h * kDH + col] = f2bf(val);
        }
    }
}

// ---------------------------------------------------------------------------
// Launch
// ---------------------------------------------------------------------------
extern "C" void kernel_launch(void* const* d_in, const int* in_sizes, int n_in,
                              void* d_out, int out_size, void* d_ws, size_t ws_size,
                              hipStream_t stream) {
    (void)in_sizes; (void)n_in; (void)out_size; (void)ws_size;

    const float* inp  = (const float*)d_in[0];
    const float* ln1s = (const float*)d_in[1];
    const float* ln1b = (const float*)d_in[2];
    const float* wq   = (const float*)d_in[3];
    const float* wk   = (const float*)d_in[4];
    const float* wv   = (const float*)d_in[5];
    const float* wo   = (const float*)d_in[6];
    const float* ln2s = (const float*)d_in[7];
    const float* ln2b = (const float*)d_in[8];
    const float* w1   = (const float*)d_in[9];
    const float* b1   = (const float*)d_in[10];
    const float* w2   = (const float*)d_in[11];
    const float* b2   = (const float*)d_in[12];
    float* out = (float*)d_out;

    char* ws = (char*)d_ws;
    size_t off = 0;
    auto alloc = [&](size_t bytes) -> char* {
        char* p = ws + off;
        off += (bytes + 255) & ~(size_t)255;
        return p;
    };
    bf16*  xh    = (bf16*)alloc((size_t)kRows * kD * 2);
    bf16*  wqkvT = (bf16*)alloc((size_t)3 * kD * kD * 2);
    bf16*  woT   = (bf16*)alloc((size_t)kD * kD * 2);
    bf16*  w1T   = (bf16*)alloc((size_t)kM * kD * 2);
    bf16*  w2T   = (bf16*)alloc((size_t)kD * kM * 2);
    bf16*  qkv   = (bf16*)alloc((size_t)3 * kRows * kD * 2);
    bf16*  Obuf  = (bf16*)alloc((size_t)kRows * kD * 2);
    float* x2    = (float*)alloc((size_t)kRows * kD * 4);
    bf16*  ybuf  = (bf16*)alloc((size_t)kRows * kD * 2);
    bf16*  hbuf  = (bf16*)alloc((size_t)kRows * kM * 2);
    int*   idx   = (int*)alloc((size_t)kNB * 64 * 4);
    int*   cnt   = (int*)alloc((size_t)kNB * 4);

    init_idx_kernel<<<1, 64, 0, stream>>>(idx, cnt);
    pack_wT_kernel<<<(kD * kD) / 256, 256, 0, stream>>>(wq, wqkvT,               kD, kD, 0.125f);
    pack_wT_kernel<<<(kD * kD) / 256, 256, 0, stream>>>(wk, wqkvT + kD * kD,     kD, kD, 1.0f);
    pack_wT_kernel<<<(kD * kD) / 256, 256, 0, stream>>>(wv, wqkvT + 2 * kD * kD, kD, kD, 1.0f);
    pack_wT_kernel<<<(kD * kD) / 256, 256, 0, stream>>>(wo, woT, kD, kD, 1.0f);
    pack_wT_kernel<<<(kD * kM) / 256, 256, 0, stream>>>(w1, w1T, kD, kM, 1.0f);
    pack_wT_kernel<<<(kM * kD) / 256, 256, 0, stream>>>(w2, w2T, kM, kD, 1.0f);

    ln_bf16_kernel<<<kRows / 8, 256, 0, stream>>>(inp, ln1s, ln1b, xh);

    gemm_bf16_kernel<0><<<dim3((3 * kD) / 128, kRows / 128), 256, 0, stream>>>(
        xh, wqkvT, 3 * kD, kD, qkv, nullptr, nullptr, nullptr);

    const bf16* qp = qkv;
    const bf16* kp = qkv + (size_t)kRows * kD;
    const bf16* vp = qkv + (size_t)2 * kRows * kD;
    bigbird_attn_kernel<<<kB * kH * kNB, 256, 0, stream>>>(qp, kp, vp, idx, cnt, Obuf);

    gemm_bf16_kernel<1><<<dim3(kD / 128, kRows / 128), 256, 0, stream>>>(
        Obuf, woT, kD, kD, nullptr, x2, inp, nullptr);

    ln_bf16_kernel<<<kRows / 8, 256, 0, stream>>>(x2, ln2s, ln2b, ybuf);

    gemm_bf16_kernel<2><<<dim3(kM / 128, kRows / 128), 256, 0, stream>>>(
        ybuf, w1T, kM, kD, hbuf, nullptr, nullptr, b1);

    gemm_bf16_kernel<3><<<dim3(kD / 128, kRows / 128), 256, 0, stream>>>(
        hbuf, w2T, kD, kM, nullptr, out, x2, b2);
}
